// LocalCAN_58729382805974
// MI455X (gfx1250) — matrix operand order repaired and checked by reading, not compiled, not run on hardware
//
#include <hip/hip_runtime.h>
#include <stdint.h>

// LocalCAN: out[s,b,:] = sum_w softmax_w( dot(sigmoid(x[s+w,b,:]*m), k[w]) ) * x[s+w,b,:]*m
// S=200, B=1024, H=128, W=3, f32. Bandwidth-bound (~230MB traffic @ 23.3TB/s ~ 10us).
// One wave32 per (batch, 14-step tile): TDM loads a 16x512B tile to LDS (zero-fill
// beyond S, padded rows for bank-conflict-free LDS), one WMMA chain computes all 16
// gate logits, fast-math sigmoid/softmax (v_exp2 + v_rcp), coalesced float4 epilogue.

#define S_DIM 200
#define B_DIM 1024
#define H_DIM 128
#define W_DIM 3

#define ROWS     16                          // rows staged + gated per tile (WMMA M)
#define TILE_OUT (ROWS - 2)                  // 14 outputs per tile (window halo = 2)
#define ROW_F    (H_DIM + 4)                 // 132 floats (528B) LDS row stride
#define TILES_PER_B ((S_DIM + TILE_OUT - 1) / TILE_OUT)   // 15

#if defined(__has_builtin)
#if __has_builtin(__builtin_amdgcn_tensor_load_to_lds)
#define HAVE_TDM 1
#endif
#endif

typedef _Float16     v16h __attribute__((ext_vector_type(16)));
typedef float        v8f  __attribute__((ext_vector_type(8)));
typedef unsigned int v4u  __attribute__((ext_vector_type(4)));
typedef int          v8i  __attribute__((ext_vector_type(8)));
typedef int          v4i  __attribute__((ext_vector_type(4)));

#define LOG2E 1.4426950408889634f

__device__ __forceinline__ float fast_rcp(float v) {
#if __has_builtin(__builtin_amdgcn_rcpf)
  return __builtin_amdgcn_rcpf(v);
#else
  return 1.0f / v;
#endif
}
__device__ __forceinline__ float fast_exp2(float v) {
#if __has_builtin(__builtin_amdgcn_exp2f)
  return __builtin_amdgcn_exp2f(v);
#else
  return exp2f(v);
#endif
}
// sigmoid(x*m) with f = -m*log2e pre-folded: rcp(1 + exp2(x*f))
__device__ __forceinline__ float sigm2(float x, float f) {
  return fast_rcp(1.0f + fast_exp2(x * f));
}

__device__ __forceinline__ float rl(float v, int lane) {
  return __int_as_float(__builtin_amdgcn_readlane(__float_as_int(v), lane));
}
__device__ __forceinline__ float bperm(float v, int srclane) {
  return __int_as_float(__builtin_amdgcn_ds_bpermute(srclane << 2, __float_as_int(v)));
}

__device__ __forceinline__ void sig4(v16h& a, int base, float4 v, float f) {
  a[base + 0] = (_Float16)sigm2(v.x, f);
  a[base + 1] = (_Float16)sigm2(v.y, f);
  a[base + 2] = (_Float16)sigm2(v.z, f);
  a[base + 3] = (_Float16)sigm2(v.w, f);
}
__device__ __forceinline__ void cvt4(v16h& a, int base, float4 v) {
  a[base + 0] = (_Float16)v.x;
  a[base + 1] = (_Float16)v.y;
  a[base + 2] = (_Float16)v.z;
  a[base + 3] = (_Float16)v.w;
}

// D (16x16 f32): VGPR r, lane n<16 -> (M=r, N=n); lane 16+n -> (M=8+r, N=n).
__device__ __forceinline__ float dget(const v8f& c, int rho, int w) {
  if (rho < 8) return rl(c[rho], w);
  return rl(c[rho - 8], 16 + w);
}

__global__ __launch_bounds__(32) void LocalCAN_kernel(
    const float* __restrict__ x, const float* __restrict__ mk,
    const float* __restrict__ kr, float* __restrict__ out)
{
  __shared__ float smem[ROWS * ROW_F + 4];
  const int lane = threadIdx.x;
  const int b    = (int)blockIdx.x / TILES_PER_B;
  const int s0   = ((int)blockIdx.x % TILES_PER_B) * TILE_OUT;

  const uint64_t gaddr = (uint64_t)(uintptr_t)(x + ((size_t)s0 * B_DIM + b) * H_DIM);
  const uint32_t lds_base = (uint32_t)(uintptr_t)(void*)&smem[0];

#ifdef HAVE_TDM
  // --- TDM 2D tile: 16 rows x 128 f32, row stride B*H, zero-fill rows >= S,
  //     LDS pad 4 dwords after every 128 dwords (row stride 528B). ---
  v4u g0;
  g0[0] = 1u;                                              // count=1, user mode
  g0[1] = lds_base;                                        // lds_addr
  g0[2] = (uint32_t)gaddr;                                 // global_addr[31:0]
  g0[3] = (uint32_t)((gaddr >> 32) & 0x01FFFFFFu) | (2u << 30);  // addr[56:32] | type=2
  v8i g1;
  g1[0] = (int)((2u << 16)        // data_size = 4B
              | (1u << 20)        // pad_enable
              | (6u << 22)        // pad_interval: 2^(6+1)=128 dwords
              | (3u << 25));      // pad_amount: 3+1=4 dwords
  g1[1] = (int)((uint32_t)H_DIM << 16);          // tensor_dim0 = 128
  g1[2] = (int)((uint32_t)(S_DIM - s0) << 16);   // tensor_dim1 = remaining rows
  g1[3] = (int)((uint32_t)H_DIM << 16);          // tile_dim0 = 128
  g1[4] = ROWS;                                  // tile_dim1 = 16, tile_dim2 = 0
  g1[5] = B_DIM * H_DIM;                         // tensor_dim0_stride = 131072
  g1[6] = 0;
  g1[7] = 0;
  v4i g2 = {0, 0, 0, 0};
  v4i g3 = {0, 0, 0, 0};
#if defined(__clang_major__) && (__clang_major__ >= 23)
  v8i g4 = {0, 0, 0, 0, 0, 0, 0, 0};
  __builtin_amdgcn_tensor_load_to_lds(g0, g1, g2, g3, g4, 0);
#else
  __builtin_amdgcn_tensor_load_to_lds(g0, g1, g2, g3, 0);
#endif
#endif // HAVE_TDM

  // --- B fragments (conv kernel, f16), 4 K-chunks of 32. B 32x16 layout:
  //     lane n = N column; halfs 0..7 -> K=kb+0..7, halfs 8..15 -> K=16+kb+0..7 (kb=0/8). ---
  const int n   = lane & 15;
  const int kbB = (lane < 16) ? 0 : 8;
  v16h bf[4];
#pragma unroll
  for (int c = 0; c < 4; ++c) {
    v16h t = {};
    if (n < W_DIM) {
      const float4* kp = (const float4*)(kr + (size_t)n * H_DIM + 32 * c + kbB);
      cvt4(t, 0, kp[0]); cvt4(t, 4, kp[1]); cvt4(t, 8, kp[4]); cvt4(t, 12, kp[5]);
    }
    bf[c] = t;
  }

  // Mask for row (s0+lane), lanes 0..15 (0 beyond S; those rows are zero anyway).
  float mval = 0.0f;
  if ((s0 + lane) < S_DIM)
    mval = mk[(size_t)(s0 + lane) * B_DIM + b];

#ifdef HAVE_TDM
  __builtin_amdgcn_s_wait_tensorcnt(0);
  __asm__ __volatile__("" ::: "memory");
#else
  // Fallback: direct loads (single wave -> DScnt ordering, no barrier needed).
  for (int r = 0; r < ROWS; ++r) {
    float4 v = make_float4(0.f, 0.f, 0.f, 0.f);
    const int s = s0 + r;
    if (s < S_DIM)
      v = *(const float4*)(x + ((size_t)s * B_DIM + b) * H_DIM + 4 * lane);
    *(float4*)(smem + (size_t)r * ROW_F + 4 * lane) = v;
  }
#endif

  // --- Gate GEMM: d[n, row] = sum_h sigmoid(x[row,h]*m) * k[n,h], rows 0..15.
  //     A 16x32 f16 layout: lane M=lane%16; lanes<16 K-base 0, lanes>=16 K-base 8;
  //     halfs 0..7 -> K=kb+0..7, halfs 8..15 -> K=16+kb+0..7. ---
  v8f acc = {0.f, 0.f, 0.f, 0.f, 0.f, 0.f, 0.f, 0.f};
  const int rm  = lane & 15;
  const int kbA = (lane < 16) ? 0 : 8;
  const float fA = bperm(mval, rm) * (-LOG2E);   // fold mask + log2e into exp2 arg
#pragma unroll
  for (int c = 0; c < 4; ++c) {
    const float* rp = smem + (size_t)rm * ROW_F + 32 * c + kbA;
    v16h a = {};
    sig4(a, 0,  *(const float4*)(rp),      fA);
    sig4(a, 4,  *(const float4*)(rp + 4),  fA);
    sig4(a, 8,  *(const float4*)(rp + 16), fA);
    sig4(a, 12, *(const float4*)(rp + 20), fA);
    acc = __builtin_amdgcn_wmma_f32_16x16x32_f16(false, a, false, bf[c],
                                                 (short)0, acc, false, false);
  }

  // --- Softmax over window + weighted sum; rolling 3-row register window. ---
  float4 r0 = *(const float4*)(smem + 0 * ROW_F + 4 * lane);
  float4 r1 = *(const float4*)(smem + 1 * ROW_F + 4 * lane);
#pragma unroll
  for (int t = 0; t < TILE_OUT; ++t) {
    float4 r2 = *(const float4*)(smem + (size_t)(t + 2) * ROW_F + 4 * lane);
    const float e0 = dget(acc, t,     0);
    const float e1 = dget(acc, t + 1, 1);
    const float e2 = dget(acc, t + 2, 2);
    const float mx = fmaxf(e0, fmaxf(e1, e2));
    const float w0 = fast_exp2((e0 - mx) * LOG2E);
    const float w1 = fast_exp2((e1 - mx) * LOG2E);
    const float w2 = fast_exp2((e2 - mx) * LOG2E);
    const float inv = fast_rcp(w0 + w1 + w2);
    const float c0 = w0 * inv * rl(mval, t);
    const float c1 = w1 * inv * rl(mval, t + 1);
    const float c2 = w2 * inv * rl(mval, t + 2);
    const int s = s0 + t;
    if (s < S_DIM) {
      float4 o;
      o.x = c0 * r0.x + c1 * r1.x + c2 * r2.x;
      o.y = c0 * r0.y + c1 * r1.y + c2 * r2.y;
      o.z = c0 * r0.z + c1 * r1.z + c2 * r2.z;
      o.w = c0 * r0.w + c1 * r1.w + c2 * r2.w;
      *(float4*)(out + ((size_t)s * B_DIM + b) * H_DIM + 4 * lane) = o;
    }
    r0 = r1;
    r1 = r2;
  }
}

extern "C" void kernel_launch(void* const* d_in, const int* in_sizes, int n_in,
                              void* d_out, int out_size, void* d_ws, size_t ws_size,
                              hipStream_t stream) {
  (void)in_sizes; (void)n_in; (void)out_size; (void)d_ws; (void)ws_size;
  const float* x  = (const float*)d_in[0];
  const float* mk = (const float*)d_in[1];
  const float* kr = (const float*)d_in[2];
  float* out = (float*)d_out;
  dim3 grid(B_DIM * TILES_PER_B);  // 1024 batches x 15 s-tiles, one wave32 each
  dim3 block(32);
  LocalCAN_kernel<<<grid, block, 0, stream>>>(x, mk, kr, out);
}
